// D_27298812134037
// MI455X (gfx1250) — matrix operand order repaired and checked by reading, not compiled, not run on hardware
//
#include <hip/hip_runtime.h>

// DST slot-status co-attention, CDNA5 (gfx1250) WMMA implementation.
// All GEMM-shaped contractions run on v_wmma_f32_16x16x32_f16 (f16 in, f32 acc).
// Requires ws_size >= ~108 MB.

typedef __attribute__((ext_vector_type(16))) _Float16 v16h;
typedef __attribute__((ext_vector_type(8)))  _Float16 v8h;
typedef __attribute__((ext_vector_type(8)))  float    v8f;

static constexpr int Bc = 64, Wc = 16, Tc = 64, Uc = 128, Sc = 20, Nc = 8;
#define NEG_INF_MASK (-100000.0f)

// ---------- WMMA fragment helpers (wave32; A and B both use contiguous k order) ----------
// A (16x32 f16): lane = hi*16 + row; halves [0..15] = K(kb + hi*16 + h)
__device__ __forceinline__ v16h make_a(const _Float16* Arow, int kb, int hi) {
  return *(const v16h*)(Arow + kb + hi * 16);
}
// B (32x16 f16) stored transposed in memory as BT[N][K]:
// lane = hi*16 + col; halves [0..15] = K(kb + hi*16 + h)  -> one aligned v16h load
__device__ __forceinline__ v16h make_b(const _Float16* BT, int K, int col, int kb, int hi) {
  return *(const v16h*)(BT + (size_t)col * K + kb + hi * 16);
}
__device__ __forceinline__ v8f wmma_step(v16h a, v16h b, v8f c) {
  return __builtin_amdgcn_wmma_f32_16x16x32_f16(false, a, false, b, (short)0, c, false, false);
}

// ---------- precision-convert kernels ----------
__global__ void k_cvt_f16(const float* __restrict__ src, _Float16* __restrict__ dst, int n) {
  int i = blockIdx.x * blockDim.x + threadIdx.x;
  if (i < n) dst[i] = (_Float16)src[i];
}
// dst[v*K + k] = src[k*N + v]  (store B operands transposed for contiguous K reads)
__global__ void k_cvt_f16_T(const float* __restrict__ src, _Float16* __restrict__ dst, int K, int N) {
  int i = blockIdx.x * blockDim.x + threadIdx.x;
  if (i < K * N) { int k = i / N, v = i % N; dst[(size_t)v * K + k] = (_Float16)src[i]; }
}

// ---------- token attention -> q_slot (f16) ----------
__global__ void __launch_bounds__(256) k_token_attn(
    const float* __restrict__ slot_utt_h, const float* __restrict__ slot_cand,
    const float* __restrict__ pos_enc, _Float16* __restrict__ q_slot_h) {
  __shared__ float Hs[Tc * Uc];  // 32 KB
  __shared__ float Cs[Sc * Uc];  // 10 KB
  __shared__ float P[Sc * Tc];   // 5 KB
  const int tid = threadIdx.x;
  const int bw = blockIdx.x;  // b*W + w
  const float* Hsrc = slot_utt_h + (size_t)bw * Tc * Uc;
  for (int i = tid; i < Tc * Uc; i += 256) Hs[i] = Hsrc[i];
  for (int i = tid; i < Sc * Uc; i += 256) Cs[i] = slot_cand[i];
  __syncthreads();
  for (int i = tid; i < Sc * Tc; i += 256) {
    int s = i / Tc, t = i % Tc;
    float acc = 0.f;
#pragma unroll 8
    for (int u = 0; u < Uc; ++u) acc += Cs[s * Uc + u] * Hs[t * Uc + u];
    P[i] = (acc == 0.0f) ? NEG_INF_MASK : acc;  // p + (-INF where p==0)
  }
  __syncthreads();
  if (tid < Sc) {
    float m = -3.4e38f;
    for (int t = 0; t < Tc; ++t) m = fmaxf(m, P[tid * Tc + t]);
    float sum = 0.f;
    for (int t = 0; t < Tc; ++t) { float e = __expf(P[tid * Tc + t] - m); P[tid * Tc + t] = e; sum += e; }
    float inv = 1.0f / sum;
    for (int t = 0; t < Tc; ++t) P[tid * Tc + t] *= inv;
  }
  __syncthreads();
  for (int i = tid; i < Sc * Uc; i += 256) {
    int s = i / Uc, u = i % Uc;
    float acc = pos_enc[(size_t)bw * Sc * Uc + i];
#pragma unroll 8
    for (int t = 0; t < Tc; ++t) acc += P[s * Tc + t] * Hs[t * Uc + u];
    q_slot_h[(size_t)bw * Sc * Uc + i] = (_Float16)acc;
  }
}

// ---------- qW = q_slot @ weight  (M=B*W*S, K=N=128) ----------
__global__ void __launch_bounds__(256) k_gemm_qw(
    const _Float16* __restrict__ A, const _Float16* __restrict__ BT, _Float16* __restrict__ C) {
  const int lane = threadIdx.x & 31, wave = threadIdx.x >> 5;
  const int hi = lane >> 4, lr = lane & 15;
  const int row0 = blockIdx.x * 16;
  const int col = wave * 16 + lr;
  const _Float16* Arow = A + (size_t)(row0 + lr) * Uc;
  v8f acc = {};
#pragma unroll
  for (int kb = 0; kb < Uc; kb += 32)
    acc = wmma_step(make_a(Arow, kb, hi), make_b(BT, Uc, col, kb, hi), acc);
#pragma unroll
  for (int r = 0; r < 8; ++r)
    C[(size_t)(row0 + r + hi * 8) * Uc + col] = (_Float16)acc[r];
}

// ---------- co[b,(w,s),(x,n)] = qW[b,(w,s),:] . q_status[b,x,n,:] ----------
__global__ void __launch_bounds__(256) k_co(
    const _Float16* __restrict__ qW, const _Float16* __restrict__ qstat, float* __restrict__ co) {
  const int lane = threadIdx.x & 31, wave = threadIdx.x >> 5;
  const int hi = lane >> 4, lr = lane & 15;
  const int b = blockIdx.x / (Wc * Sc / 16);  // 20 row-tiles per batch
  const int tile = blockIdx.x % (Wc * Sc / 16);
  const int row0 = b * (Wc * Sc) + tile * 16;
  const int col = wave * 16 + lr;  // (x,n) = x*8+n
  const _Float16* Arow = qW + (size_t)(row0 + lr) * Uc;
  const _Float16* BT = qstat + (size_t)b * (Wc * Nc) * Uc;  // [col=(x,n)][k=u]
  v8f acc = {};
#pragma unroll
  for (int kb = 0; kb < Uc; kb += 32)
    acc = wmma_step(make_a(Arow, kb, hi), make_b(BT, Uc, col, kb, hi), acc);
#pragma unroll
  for (int r = 0; r < 8; ++r)
    co[(size_t)(row0 + r + hi * 8) * (Wc * Nc) + col] = acc[r];
}

// ---------- softmax over x + weighted status sum -> q_status_slot (f16) ----------
__global__ void __launch_bounds__(256) k_p2_wsum(
    const float* __restrict__ co, const float* __restrict__ q_status, _Float16* __restrict__ qss) {
  const int lane = threadIdx.x & 31, wave = threadIdx.x >> 5;
  const int r = blockIdx.x * 8 + wave;  // row (b,w,s,n), total B*W*S*N
  const int n = r & 7;
  const int t = r >> 3;        // (b*W+w)*S + s  == co row
  const int bw = t / Sc;
  const int b = bw >> 4;
  const float* cv = co + (size_t)t * (Wc * Nc) + n;
  float v[Wc];
  float m = -3.4e38f;
#pragma unroll
  for (int x = 0; x < Wc; ++x) {
    float c = cv[x * Nc];
    c = (c == 0.0f) ? NEG_INF_MASK : c;
    v[x] = c;
    m = fmaxf(m, c);
  }
  float sum = 0.f;
#pragma unroll
  for (int x = 0; x < Wc; ++x) { v[x] = __expf(v[x] - m); sum += v[x]; }
  float inv = 1.f / sum;
#pragma unroll
  for (int x = 0; x < Wc; ++x) v[x] *= inv;
#pragma unroll
  for (int j = 0; j < 4; ++j) {
    int u = lane + j * 32;
    float acc = 0.f;
#pragma unroll
    for (int x = 0; x < Wc; ++x)
      acc += v[x] * q_status[(((size_t)b * Wc + x) * Nc + n) * Uc + u];
    qss[(size_t)r * Uc + u] = (_Float16)acc;
  }
}

// ---------- MLP layer 1: relu([q_slot|q_status_slot] @ W1 + b1), concat via split K loop ----------
__global__ void __launch_bounds__(256) k_mlp1(
    const _Float16* __restrict__ qslot, const _Float16* __restrict__ qss,
    const _Float16* __restrict__ W1t, const float* __restrict__ b1, _Float16* __restrict__ h1) {
  const int lane = threadIdx.x & 31, wave = threadIdx.x >> 5;
  const int hi = lane >> 4, lr = lane & 15;
  const int row0 = blockIdx.x * 16;
  const int col = wave * 16 + lr;
  const _Float16* Aq = qslot + (size_t)((row0 + lr) >> 3) * Uc;  // q_slot row = r / N
  const _Float16* As = qss + (size_t)(row0 + lr) * Uc;
  v8f acc = {};
#pragma unroll
  for (int kb = 0; kb < Uc; kb += 32)
    acc = wmma_step(make_a(Aq, kb, hi), make_b(W1t, 2 * Uc, col, kb, hi), acc);
#pragma unroll
  for (int kb = 0; kb < Uc; kb += 32)
    acc = wmma_step(make_a(As, kb, hi), make_b(W1t, 2 * Uc, col, Uc + kb, hi), acc);
  const float bias = b1[col];
#pragma unroll
  for (int r = 0; r < 8; ++r) {
    float vv = fmaxf(acc[r] + bias, 0.f);
    h1[(size_t)(row0 + r + hi * 8) * Uc + col] = (_Float16)vv;
  }
}

// ---------- MLP layer 2 fused with W3 projection -> logits ----------
__global__ void __launch_bounds__(256) k_mlp2(
    const _Float16* __restrict__ h1, const _Float16* __restrict__ W2t,
    const float* __restrict__ b2, const float* __restrict__ W3,
    const float* __restrict__ b3, float* __restrict__ logits) {
  __shared__ float slog[16];
  const int tid = threadIdx.x;
  const int lane = tid & 31, wave = tid >> 5;
  const int hi = lane >> 4, lr = lane & 15;
  const int row0 = blockIdx.x * 16;
  const int col = wave * 16 + lr;
  if (tid < 16) slog[tid] = 0.f;
  __syncthreads();
  const _Float16* Arow = h1 + (size_t)(row0 + lr) * Uc;
  v8f acc = {};
#pragma unroll
  for (int kb = 0; kb < Uc; kb += 32)
    acc = wmma_step(make_a(Arow, kb, hi), make_b(W2t, Uc, col, kb, hi), acc);
  const float bias = b2[col];
  const float w3 = W3[col];
#pragma unroll
  for (int r = 0; r < 8; ++r) {
    float vv = fmaxf(acc[r] + bias, 0.f);
    atomicAdd(&slog[r + hi * 8], vv * w3);  // ds_add_f32 row reduction
  }
  __syncthreads();
  if (tid < 16) logits[row0 + tid] = slog[tid] + b3[0];
}

// ---------- masked max-pool over w + label + output packing ----------
__global__ void k_finalize(const float* __restrict__ logits, const float* __restrict__ mask,
                           float* __restrict__ out) {
  const int b = blockIdx.x, j = threadIdx.x;
  if (j >= Sc * Nc) return;
  float m = -3.4e38f;
  for (int w = 0; w < Wc; ++w) {
    size_t idx = ((size_t)b * Wc + w) * (Sc * Nc) + j;
    m = fmaxf(m, logits[idx] + mask[idx]);
  }
  out[(size_t)b * (Sc * Nc) + j] = (m > 0.f) ? 1.f : 0.f;                   // labels
  out[(size_t)(Bc * Sc * Nc) + 1 + (size_t)b * (Sc * Nc) + j] = m;          // logits
  if (b == 0 && j == 0) out[Bc * Sc * Nc] = (float)(Sc * Nc);               // S*N scalar
}

extern "C" void kernel_launch(void* const* d_in, const int* in_sizes, int n_in,
                              void* d_out, int out_size, void* d_ws, size_t ws_size,
                              hipStream_t stream) {
  (void)in_sizes; (void)n_in; (void)out_size; (void)ws_size;
  const float* slot_utt_h = (const float*)d_in[0];
  const float* slot_cand  = (const float*)d_in[1];
  // d_in[2] status_candidate_c is unused by the reference
  const float* pos_enc    = (const float*)d_in[3];
  const float* q_status   = (const float*)d_in[4];
  const float* mask       = (const float*)d_in[5];
  const float* weight     = (const float*)d_in[6];
  const float* W1         = (const float*)d_in[7];
  const float* b1         = (const float*)d_in[8];
  const float* W2         = (const float*)d_in[9];
  const float* b2         = (const float*)d_in[10];
  const float* W3         = (const float*)d_in[11];
  const float* b3         = (const float*)d_in[12];

  // workspace layout (256-B aligned offsets), total ~107.7 MB
  char* ws = (char*)d_ws;
  _Float16* qslot_h  = (_Float16*)(ws + 0);           // B*W*S*U f16   = 5,242,880 B
  _Float16* qW_h     = (_Float16*)(ws + 5242880);     // B*W*S*U f16
  _Float16* qstat_h  = (_Float16*)(ws + 10485760);    // B*W*N*U f16   = 2,097,152 B
  _Float16* weight_t = (_Float16*)(ws + 12582912);    // 128x128 f16
  _Float16* W1t      = (_Float16*)(ws + 12615680);    // 128x256 f16
  _Float16* W2t      = (_Float16*)(ws + 12681216);    // 128x128 f16
  float*    co       = (float*)   (ws + 12713984);    // B*W*S * W*N f32 = 10,485,760 B
  _Float16* qss_h    = (_Float16*)(ws + 23199744);    // B*W*S*N*U f16 = 41,943,040 B
  _Float16* h1_h     = (_Float16*)(ws + 65142784);    // B*W*S*N*U f16 = 41,943,040 B
  float*    logits   = (float*)   (ws + 107085824);   // B*W*S*N f32   = 655,360 B

  const int MR = Bc * Wc * Sc * Nc;  // 163840 MLP rows

  k_cvt_f16<<<(Bc * Wc * Nc * Uc + 255) / 256, 256, 0, stream>>>(q_status, qstat_h, Bc * Wc * Nc * Uc);
  k_cvt_f16_T<<<(Uc * Uc + 255) / 256, 256, 0, stream>>>(weight, weight_t, Uc, Uc);
  k_cvt_f16_T<<<(2 * Uc * Uc + 255) / 256, 256, 0, stream>>>(W1, W1t, 2 * Uc, Uc);
  k_cvt_f16_T<<<(Uc * Uc + 255) / 256, 256, 0, stream>>>(W2, W2t, Uc, Uc);

  k_token_attn<<<Bc * Wc, 256, 0, stream>>>(slot_utt_h, slot_cand, pos_enc, qslot_h);
  k_gemm_qw<<<(Bc * Wc * Sc) / 16, 256, 0, stream>>>(qslot_h, weight_t, qW_h);
  k_co<<<Bc * (Wc * Sc / 16), 256, 0, stream>>>(qW_h, qstat_h, co);
  k_p2_wsum<<<MR / 8, 256, 0, stream>>>(co, q_status, qss_h);
  k_mlp1<<<MR / 16, 256, 0, stream>>>(qslot_h, qss_h, W1t, b1, h1_h);
  k_mlp2<<<MR / 16, 256, 0, stream>>>(h1_h, W2t, b2, W3, b3, logits);
  k_finalize<<<Bc, Sc * Nc, 0, stream>>>(logits, mask, (float*)d_out);
}
